// AttnAdapter_75660143886542
// MI455X (gfx1250) — matrix operand order repaired and checked by reading, not compiled, not run on hardware
//
#include <hip/hip_runtime.h>
#include <cstddef>
#include <cstdint>

// ---------------------------------------------------------------------------
// Types for CDNA5 WMMA (wave32): v_wmma_f32_16x16x32_bf16
// ---------------------------------------------------------------------------
typedef __attribute__((ext_vector_type(16))) __bf16 bf16x16;
typedef __attribute__((ext_vector_type(8)))  __bf16 bf16x8;
typedef __attribute__((ext_vector_type(8)))  float  f32x8;
typedef __attribute__((ext_vector_type(4)))  float  f32x4;
typedef __attribute__((ext_vector_type(4)))  unsigned int u32x4;
typedef __attribute__((ext_vector_type(8)))  int          i32x8;
typedef __attribute__((ext_vector_type(4)))  int          i32x4;

__device__ __forceinline__ bf16x16 cat8(bf16x8 lo, bf16x8 hi) {
  return __builtin_shufflevector(lo, hi, 0,1,2,3,4,5,6,7,8,9,10,11,12,13,14,15);
}

__device__ __forceinline__ f32x8 wmma_bf16(bf16x16 a, bf16x16 b, f32x8 c) {
  // 8 args: (neg_a, A, neg_b, B, c_mod, C, reuse_a, reuse_b)
  return __builtin_amdgcn_wmma_f32_16x16x32_bf16(false, a, false, b, (short)0, c,
                                                 false, false);
}

// ---------------------------------------------------------------------------
// Tensor Data Mover: async 2D bf16 tile load Global -> LDS (TENSORcnt tracked).
// D# per CDNA5 ISA ch.8: group0 = {count/type/addr}, group1 = {dims/tile/stride}.
// This toolchain (clang-23/therock-10.0) uses the 6-arg builtin:
//   (u32x4 g0, i32x8 g1, i32x4 g2, i32x4 g3, i32x8 g4, i32 cpol)
// ---------------------------------------------------------------------------
__device__ __forceinline__ void tdm_load_2d_bf16(uint32_t lds_off,
                                                 const __bf16* gptr,
                                                 uint32_t tensor_w,  // elems/row
                                                 uint32_t tensor_h,  // rows
                                                 uint32_t tile_w,
                                                 uint32_t tile_h) {
  const unsigned long long ga = (unsigned long long)(uintptr_t)gptr;
  u32x4 g0;
  g0[0] = 1u;                                               // count=1 (valid)
  g0[1] = lds_off;                                          // lds_addr (bytes)
  g0[2] = (unsigned)(ga & 0xffffffffu);                     // global_addr[31:0]
  g0[3] = (unsigned)((ga >> 32) & 0x01ffffffu) | (2u << 30);// addr[56:32]|type=2
  i32x8 g1;
  g1[0] = (int)(1u << 16);                                  // data_size=1 (2B)
  g1[1] = (int)((tensor_w & 0xffffu) << 16);                // tensor_dim0 lo16
  g1[2] = (int)((tensor_w >> 16) | ((tensor_h & 0xffffu) << 16)); // d0hi|d1lo
  g1[3] = (int)((tensor_h >> 16) | (tile_w << 16));         // d1 hi | tile_dim0
  g1[4] = (int)(tile_h & 0xffffu);                          // tile_dim1 (dim2=0)
  g1[5] = (int)tensor_w;                                    // dim0_stride lo32
  g1[6] = 0;
  g1[7] = 0;
  const i32x4 gz4 = {0, 0, 0, 0};                           // 2D: groups 2/3 off
  const i32x8 gz8 = {0, 0, 0, 0, 0, 0, 0, 0};
  __builtin_amdgcn_tensor_load_to_lds(g0, g1, gz4, gz4, gz8, 0);
}

__device__ __forceinline__ bf16x8 ld8cvt(const float* p) {
  f32x4 a = *(const f32x4*)p;
  f32x4 b = *(const f32x4*)(p + 4);
  bf16x8 r;
#pragma unroll
  for (int i = 0; i < 4; ++i) { r[i] = (__bf16)a[i]; r[4 + i] = (__bf16)b[i]; }
  return r;
}

// ---------------------------------------------------------------------------
// Generic tiled GEMM: C[M,N] (f32) = A[M,K] (f32 or bf16) * B[K,N] (f32)
// Block tile 128x128, BK=32, 256 threads = 8 waves, wave tile 32x64 (2x4 WMMA)
// bf16-A path stages the A tile via the Tensor Data Mover.
// ---------------------------------------------------------------------------
template <typename AT>
__global__ __launch_bounds__(256) void gemm_wmma(const AT* __restrict__ A,
                                                 const float* __restrict__ B,
                                                 float* __restrict__ C,
                                                 int M, int N, int K) {
  constexpr int BM = 128, BN = 128, BK = 32;
  constexpr bool A_BF16 = (sizeof(AT) == 2);
  __shared__ __bf16 As[BM][BK];   // row-major (m, k)      8 KB
  __shared__ __bf16 Bs[BN][BK];   // transposed (n, k)     8 KB

  const int tid  = threadIdx.x;
  const int bm0  = blockIdx.y * BM;
  const int bn0  = blockIdx.x * BN;
  const int wid  = tid >> 5;
  const int lane = tid & 31;
  const int wm   = wid & 3;        // 0..3 -> 32-row strip
  const int wn   = wid >> 2;       // 0..1 -> 64-col strip
  const int half = lane >> 4;      // lane group
  const int l16  = lane & 15;
  const int kcA  = half * 16;      // A: contiguous K half per lane group (kc, kc+8)

  const f32x8 zero = {};
  f32x8 acc[2][4];
#pragma unroll
  for (int i = 0; i < 2; ++i)
#pragma unroll
    for (int j = 0; j < 4; ++j) acc[i][j] = zero;

  // staging thread mapping
  const int arow = tid >> 1, acb = (tid & 1) * 16;   // A: 128 rows x 32
  const int bkr  = tid >> 3, bnb = (tid & 7) * 16;   // B: 32 rows x 128

  for (int k0 = 0; k0 < K; k0 += BK) {
    // --- stage A tile ---
    if (A_BF16) {
      if (tid < 32)   // wave 0 issues the DMA (EXEC-independent, TENSORcnt)
        tdm_load_2d_bf16((uint32_t)(uintptr_t)&As[0][0],
                         (const __bf16*)(const void*)A + (size_t)bm0 * K + k0,
                         (uint32_t)K, (uint32_t)M, BK, BM);
    } else {
      const AT* g = A + (size_t)(bm0 + arow) * K + k0 + acb;
      *(bf16x8*)&As[arow][acb]     = ld8cvt((const float*)g);
      *(bf16x8*)&As[arow][acb + 8] = ld8cvt((const float*)(g + 8));
    }
    // --- stage B tile transposed (n, k), f32 -> bf16 ---
    {
      const float* g = B + (size_t)(k0 + bkr) * N + bn0 + bnb;
#pragma unroll
      for (int u = 0; u < 16; ++u) Bs[bnb + u][bkr] = (__bf16)g[u];
      if (k0 + BK < K)  // pull next B tile toward GL2 while we compute
        __builtin_prefetch(B + (size_t)(k0 + BK + bkr) * N + bn0 + bnb, 0, 1);
    }
    if (A_BF16 && tid < 32) __builtin_amdgcn_s_wait_tensorcnt(0);
    __syncthreads();

    bf16x16 af[2], bfr[4];
#pragma unroll
    for (int i = 0; i < 2; ++i) {
      const __bf16* p = &As[wm * 32 + i * 16 + l16][0];
      af[i] = cat8(*(const bf16x8*)(p + kcA), *(const bf16x8*)(p + kcA + 8));
    }
#pragma unroll
    for (int j = 0; j < 4; ++j) {
      // B fragment: lane holds column n, contiguous K half per lane group
      bfr[j] = *(const bf16x16*)&Bs[wn * 64 + j * 16 + l16][half * 16];
    }
#pragma unroll
    for (int i = 0; i < 2; ++i)
#pragma unroll
      for (int j = 0; j < 4; ++j) acc[i][j] = wmma_bf16(af[i], bfr[j], acc[i][j]);

    __syncthreads();
  }

  // --- epilogue: C/D layout m = half*8 + r, n = l16 ---
#pragma unroll
  for (int i = 0; i < 2; ++i) {
#pragma unroll
    for (int j = 0; j < 4; ++j) {
      const int n = bn0 + wn * 64 + j * 16 + l16;
#pragma unroll
      for (int r = 0; r < 8; ++r) {
        const int m = bm0 + wm * 32 + i * 16 + half * 8 + r;
        C[(size_t)m * N + n] = acc[i][j][r];
      }
    }
  }
}

// ---------------------------------------------------------------------------
// RoPE + transpose to head-major bf16 [nh][S][128]; 1/sqrt(d) folded into Q
// ---------------------------------------------------------------------------
__global__ __launch_bounds__(128) void rope_convert(const float* __restrict__ src,
                                                    const float* __restrict__ cosb,
                                                    const float* __restrict__ sinb,
                                                    __bf16* __restrict__ dst,
                                                    int nh, float scale) {
  const int s = blockIdx.x, h = blockIdx.y, d = threadIdx.x;
  const int W = nh * 128;
  const size_t base = (size_t)s * W + h * 128;
  const float x  = src[base + d];
  const float xr = (d < 64) ? -src[base + d + 64] : src[base + d - 64];
  const float v  = (x * cosb[s * 128 + d] + xr * sinb[s * 128 + d]) * scale;
  dst[((size_t)h * 2048 + s) * 128 + d] = (__bf16)v;
}

__global__ __launch_bounds__(128) void v_convert(const float* __restrict__ src,
                                                 __bf16* __restrict__ dst) {
  const int s = blockIdx.x, h = blockIdx.y, d = threadIdx.x;
  dst[((size_t)h * 2048 + s) * 128 + d] =
      (__bf16)src[(size_t)s * 1024 + h * 128 + d];
}

// ---------------------------------------------------------------------------
// Flash-style GQA attention. 1 block = (64-row q tile, head). 4 waves x 16 rows.
// K tiles arrive via double-buffered TDM DMA (overlaps the WMMA stream);
// S = Q K^T and O += P V both via v_wmma_f32_16x16x32_bf16.
// ---------------------------------------------------------------------------
__global__ __launch_bounds__(128) void attn_kernel(const __bf16* __restrict__ Qb,
                                                   const __bf16* __restrict__ Kb,
                                                   const __bf16* __restrict__ Vb,
                                                   __bf16* __restrict__ Ob) {
  constexpr int S = 2048, HD = 128;
  constexpr int SYS = 35, BND = 611;       // SYS+IMG = 611
  __shared__ __bf16 Ks[2][64][HD];         // double-buffered K   32 KB
  __shared__ __bf16 Vs[HD][64];            // V transposed        16 KB
  __shared__ __bf16 Ps[4][16][64];         // per-wave P           8 KB

  const int qt  = blockIdx.x;              // 0..31
  const int h   = blockIdx.y;              // 0..31
  const int kvh = h >> 2;                  // GQA groups = 4
  const int tid = threadIdx.x;
  const int w    = tid >> 5;
  const int lane = tid & 31;
  const int half = lane >> 4, l16 = lane & 15;
  const int qrow0 = qt * 64 + w * 16;

  const __bf16* kbase = Kb + (size_t)kvh * S * HD;

  // Q fragments (A layout), held in registers for all K tiles
  bf16x16 aq[4];
  {
    const __bf16* qg = Qb + ((size_t)h * S + qrow0 + l16) * HD;
#pragma unroll
    for (int c = 0; c < 4; ++c)
      aq[c] = cat8(*(const bf16x8*)(qg + c * 32 + half * 16),
                   *(const bf16x8*)(qg + c * 32 + half * 16 + 8));
  }

  const f32x8 zero = {};
  f32x8 o[8];
#pragma unroll
  for (int nj = 0; nj < 8; ++nj) o[nj] = zero;
  float mstat[8], lsum[8];
#pragma unroll
  for (int r = 0; r < 8; ++r) { mstat[r] = -3.0e38f; lsum[r] = 0.0f; }

  const int ldr = tid >> 1, ldb = (tid & 1) * 64;  // cooperative load mapping

  // Pipeline prologue: DMA K tile 0 into buffer 0 (wave 0 issues).
  if (tid < 32)
    tdm_load_2d_bf16((uint32_t)(uintptr_t)&Ks[0][0][0], kbase, HD, S, HD, 64);

  for (int kt = 0; kt <= qt; ++kt) {
    const int cur = kt & 1;
    // Issue the DMA for the next K tile into the other buffer, then wait for
    // the current tile only (tensorcnt<=1): next-tile DMA overlaps compute.
    if (tid < 32) {
      if (kt < qt) {
        tdm_load_2d_bf16((uint32_t)(uintptr_t)&Ks[cur ^ 1][0][0],
                         kbase + (size_t)(kt + 1) * 64 * HD, HD, S, HD, 64);
        __builtin_amdgcn_s_wait_tensorcnt(1);
      } else {
        __builtin_amdgcn_s_wait_tensorcnt(0);
      }
    }
    // ---- stage V transposed (cooperative; TDM cannot transpose) ----
    {
      const __bf16* vg = Vb + ((size_t)kvh * S + kt * 64 + ldr) * HD + ldb;
#pragma unroll
      for (int u = 0; u < 64; ++u) Vs[ldb + u][ldr] = vg[u];
    }
    __syncthreads();

    // ---- S = Q K^T : 4 col-tiles x 4 K-chunks ----
    f32x8 sv[4];
#pragma unroll
    for (int j = 0; j < 4; ++j) {
      sv[j] = zero;
      const __bf16* kp = &Ks[cur][j * 16 + l16][0];
#pragma unroll
      for (int c = 0; c < 4; ++c) {
        bf16x16 bk = *(const bf16x16*)(kp + c * 32 + half * 16);
        sv[j] = wmma_bf16(aq[c], bk, sv[j]);
      }
    }

    // ---- mask + scale + online softmax (rows = half*8+r, cols = j*16+l16) ----
#pragma unroll
    for (int r = 0; r < 8; ++r) {
      const int qi = qrow0 + half * 8 + r;
      float loc = -3.0e38f;
#pragma unroll
      for (int j = 0; j < 4; ++j) {
        const int ki = kt * 64 + j * 16 + l16;
        float x = sv[j][r];
        if (qi >= BND) {
          if (ki < SYS)      x *= 0.1f;   // SUP
          else if (ki < BND) x *= 2.0f;   // ENH
        }
        x = (ki <= qi) ? x : -1000000000.0f;
        sv[j][r] = x;
        loc = fmaxf(loc, x);
      }
#pragma unroll
      for (int msk = 1; msk < 16; msk <<= 1)
        loc = fmaxf(loc, __shfl_xor(loc, msk, 32));
      const float nmax  = fmaxf(mstat[r], loc);
      const float alpha = __expf(mstat[r] - nmax);
      float psum = 0.0f;
#pragma unroll
      for (int j = 0; j < 4; ++j) {
        const float p = __expf(sv[j][r] - nmax);
        sv[j][r] = p;
        psum += p;
      }
#pragma unroll
      for (int msk = 1; msk < 16; msk <<= 1) psum += __shfl_xor(psum, msk, 32);
      lsum[r]  = lsum[r] * alpha + psum;
      mstat[r] = nmax;
#pragma unroll
      for (int nj = 0; nj < 8; ++nj) o[nj][r] *= alpha;
    }

    // ---- P -> LDS to re-lay into A format ----
#pragma unroll
    for (int r = 0; r < 8; ++r)
#pragma unroll
      for (int j = 0; j < 4; ++j)
        Ps[w][half * 8 + r][j * 16 + l16] = (__bf16)sv[j][r];
    __syncthreads();

    // ---- O += P V : 8 d-tiles x 2 K-chunks ----
    bf16x16 pa[2];
    {
      const __bf16* pp = &Ps[w][l16][0];
#pragma unroll
      for (int c2 = 0; c2 < 2; ++c2)
        pa[c2] = cat8(*(const bf16x8*)(pp + c2 * 32 + half * 16),
                      *(const bf16x8*)(pp + c2 * 32 + half * 16 + 8));
    }
#pragma unroll
    for (int nj = 0; nj < 8; ++nj) {
      const __bf16* vp = &Vs[nj * 16 + l16][0];
#pragma unroll
      for (int c2 = 0; c2 < 2; ++c2) {
        bf16x16 bv = *(const bf16x16*)(vp + c2 * 32 + half * 16);
        o[nj] = wmma_bf16(pa[c2], bv, o[nj]);
      }
    }
    __syncthreads();   // protect Vs/Ps (and K buffer rotation) before next tile
  }

  // ---- finalize: O /= l, write bf16 [s][h*128+d] for the output GEMM ----
#pragma unroll
  for (int r = 0; r < 8; ++r) {
    const int qi = qrow0 + half * 8 + r;
    const float inv = 1.0f / lsum[r];
#pragma unroll
    for (int nj = 0; nj < 8; ++nj) {
      const int d = nj * 16 + l16;
      Ob[(size_t)qi * 4096 + h * 128 + d] = (__bf16)(o[nj][r] * inv);
    }
  }
}

// ---------------------------------------------------------------------------
// Host-side orchestration
// ---------------------------------------------------------------------------
extern "C" void kernel_launch(void* const* d_in, const int* in_sizes, int n_in,
                              void* d_out, int out_size, void* d_ws, size_t ws_size,
                              hipStream_t stream) {
  (void)in_sizes; (void)n_in; (void)out_size; (void)ws_size;
  const float* X    = (const float*)d_in[0];
  const float* cosb = (const float*)d_in[1];
  const float* sinb = (const float*)d_in[2];
  const float* Wq   = (const float*)d_in[3];
  const float* Wk   = (const float*)d_in[4];
  const float* Wv   = (const float*)d_in[5];
  const float* Wo   = (const float*)d_in[6];
  float* out = (float*)d_out;

  char* ws = (char*)d_ws;
  // layout (bytes): Qf32 [0,32M) | Kf32 [32M,40M) | Vf32 [40M,48M)
  //                 Qbf [48M,64M) | Kbf [64M,68M) | Vbf [68M,72M)
  //                 Obf reuses [0,16M) after Q has been rope-converted.
  float*  Qf = (float*)(ws);
  float*  Kf = (float*)(ws + (size_t)33554432);
  float*  Vf = (float*)(ws + (size_t)41943040);
  __bf16* Qb = (__bf16*)(ws + (size_t)50331648);
  __bf16* Kb = (__bf16*)(ws + (size_t)67108864);
  __bf16* Vb = (__bf16*)(ws + (size_t)71303168);
  __bf16* Ob = (__bf16*)(ws);   // reuse Qf region

  const float scaling = 0.08838834764831845f;  // 128^-0.5, folded into Q

  // QKV projections (f32 A converted in-tile to bf16, WMMA f32 accum)
  gemm_wmma<float><<<dim3(32, 16), 256, 0, stream>>>(X, Wq, Qf, 2048, 4096, 4096);
  gemm_wmma<float><<<dim3(8, 16),  256, 0, stream>>>(X, Wk, Kf, 2048, 1024, 4096);
  gemm_wmma<float><<<dim3(8, 16),  256, 0, stream>>>(X, Wv, Vf, 2048, 1024, 4096);

  // RoPE (+ scale fold into Q) and head-major bf16 repack
  rope_convert<<<dim3(2048, 32), 128, 0, stream>>>(Qf, cosb, sinb, Qb, 32, scaling);
  rope_convert<<<dim3(2048, 8),  128, 0, stream>>>(Kf, cosb, sinb, Kb, 8, 1.0f);
  v_convert<<<dim3(2048, 8),     128, 0, stream>>>(Vf, Vb);

  // GQA flash attention with ENH/SUP score scaling + causal mask
  attn_kernel<<<dim3(32, 32), 128, 0, stream>>>(Qb, Kb, Vb, Ob);

  // Output projection (bf16 A path, A tile staged via TDM)
  gemm_wmma<__bf16><<<dim3(32, 16), 256, 0, stream>>>(Ob, Wo, out, 2048, 4096, 4096);
}
